// QLORA_38500086841722
// MI455X (gfx1250) — matrix operand order repaired and checked by reading, not compiled, not run on hardware
//
#include <hip/hip_runtime.h>

// ---------------------------------------------------------------------------
// Problem constants (from reference): x (4,2048,4096) f32, W (4096,4096) f32,
// A (16,4096) f32, B (4096,16) f32.  out = x @ (W + 2*B@A)^T, fp32.
// ---------------------------------------------------------------------------
#define MDIM 8192       // 4*2048
#define NDIM 4096       // Dout
#define KDIM 4096       // Din
#define RANK 16
#define SCALING 2.0f

#define BM 128
#define BN 128
#define BK 32

typedef __attribute__((ext_vector_type(16))) __bf16 v16bf;
typedef __attribute__((ext_vector_type(8)))  float  v8f;

// ---------------------------------------------------------------------------
// Kernel 1: W_eff = W_base + SCALING * (B @ A), split into bf16 hi/lo planes.
// Layout [N][K], k contiguous (same as W_base).  Block: 16 o-rows x 256 d-cols.
// ---------------------------------------------------------------------------
__global__ void __launch_bounds__(256)
qlora_prep_w(const float* __restrict__ Wb, const float* __restrict__ A,
             const float* __restrict__ Bm,
             __bf16* __restrict__ wh, __bf16* __restrict__ wl) {
  __shared__ float sA[RANK][256];
  __shared__ float sB[16][RANK];
  const int t  = threadIdx.x;
  const int d0 = blockIdx.x * 256;
  const int o0 = blockIdx.y * 16;
#pragma unroll
  for (int r = 0; r < RANK; ++r) sA[r][t] = A[r * KDIM + d0 + t];
  sB[t >> 4][t & 15] = Bm[(o0 + (t >> 4)) * RANK + (t & 15)];
  __syncthreads();
  const int d = d0 + t;
#pragma unroll 4
  for (int o = 0; o < 16; ++o) {
    float acc = 0.f;
#pragma unroll
    for (int r = 0; r < RANK; ++r) acc += sB[o][r] * sA[r][t];
    const size_t idx = (size_t)(o0 + o) * KDIM + d;
    const float v = Wb[idx] + SCALING * acc;
    const __bf16 h = (__bf16)v;
    wh[idx] = h;
    wl[idx] = (__bf16)(v - (float)h);
  }
}

// ---------------------------------------------------------------------------
// Kernel 2: split x (f32) into bf16 hi/lo planes.  8 elements / thread,
// packed 128-bit stores.
// ---------------------------------------------------------------------------
__global__ void __launch_bounds__(256)
qlora_prep_x(const float* __restrict__ x, __bf16* __restrict__ xh,
             __bf16* __restrict__ xl) {
  const size_t i = ((size_t)blockIdx.x * 256 + threadIdx.x) * 8;
  const float4 v0 = *(const float4*)(x + i);
  const float4 v1 = *(const float4*)(x + i + 4);
  float f[8] = {v0.x, v0.y, v0.z, v0.w, v1.x, v1.y, v1.z, v1.w};
  union { __bf16 h[8]; uint4 u; } H, L;
#pragma unroll
  for (int j = 0; j < 8; ++j) {
    H.h[j] = (__bf16)f[j];
    L.h[j] = (__bf16)(f[j] - (float)H.h[j]);
  }
  *(uint4*)(xh + i) = H.u;
  *(uint4*)(xl + i) = L.u;
}

// ---------------------------------------------------------------------------
// Kernel 3: C[M][N] = Xsplit @ Wsplit^T via v_wmma_f32_16x16x32_bf16 (bf16x3).
// 128x128x32 tiles, double-buffered LDS, 8 waves (2 M x 4 N), wave tile 64x32.
// ---------------------------------------------------------------------------
__global__ void __launch_bounds__(256)
qlora_gemm(const __bf16* __restrict__ xh, const __bf16* __restrict__ xl,
           const __bf16* __restrict__ wh, const __bf16* __restrict__ wl,
           float* __restrict__ out) {
  __shared__ __bf16 sXh[2][BM][BK];   // 16 KB
  __shared__ __bf16 sXl[2][BM][BK];   // 16 KB
  __shared__ __bf16 sWh[2][BN][BK];   // 16 KB  (layout [n][k], k contiguous)
  __shared__ __bf16 sWl[2][BN][BK];   // 16 KB

  const int tid  = threadIdx.x;
  const int lane = tid & 31;
  const int wave = tid >> 5;
  const int lr   = lane & 15;
  const int kh   = lane >> 4;           // K-half selector for fragments
  const int wm   = (wave & 1) * 64;     // wave M offset inside block tile
  const int wn   = (wave >> 1) * 32;    // wave N offset inside block tile
  const int bm   = blockIdx.y * BM;
  const int bn   = blockIdx.x * BN;

  // Global->reg->LDS staging map: tile is BM x BK bf16 = 512 16-byte chunks;
  // each thread moves 2 chunks per plane.
  const int c0 = tid, c1 = tid + 256;
  const int gr0 = c0 >> 2, gq0 = (c0 & 3) * 8;
  const int gr1 = c1 >> 2, gq1 = (c1 & 3) * 8;

  v8f acc[4][2] = {};          // 4 m-frags x 2 n-frags of 16x16 f32

  uint4 st[8];                 // staged chunks: xh,xh, xl,xl, wh,wh, wl,wl

  auto ldg = [&](const __bf16* __restrict__ g, int rowbase, int kt, uint4* d) {
    d[0] = *(const uint4*)(g + (size_t)(rowbase + gr0) * KDIM + kt * BK + gq0);
    d[1] = *(const uint4*)(g + (size_t)(rowbase + gr1) * KDIM + kt * BK + gq1);
  };
  auto sts = [&](__bf16* s, const uint4* d) {
    *(uint4*)(s + gr0 * BK + gq0) = d[0];
    *(uint4*)(s + gr1 * BK + gq1) = d[1];
  };
  // A fragment (16x32 bf16, M x K): lane<16 row=m0+lr holds K = 8*kh..+7 and
  // 16+8*kh..+7 (ISA 7.12.2, 16-bit A layout).  Two ds_load_b128.
  auto lfragA = [&](const __bf16* s, int m0) -> v16bf {
    union { uint4 q[2]; v16bf v; } u;
    const __bf16* p = s + (m0 + lr) * BK + 8 * kh;
    u.q[0] = *(const uint4*)p;
    u.q[1] = *(const uint4*)(p + 16);
    return u.v;
  };
  // B fragment (32x16 bf16, K x N): lane holds col n0+lr, K = 8*kh..+7 and
  // 16+8*kh..+7 in [n][k] LDS layout (matches A).  Two ds_load_b128.
  auto lfragB = [&](const __bf16* s, int n0) -> v16bf {
    union { uint4 q[2]; v16bf v; } u;
    const __bf16* p = s + (n0 + lr) * BK + 8 * kh;
    u.q[0] = *(const uint4*)p;
    u.q[1] = *(const uint4*)(p + 16);
    return u.v;
  };

  // Prologue: stage k-tile 0 into LDS buffer 0.
  ldg(xh, bm, 0, &st[0]); ldg(xl, bm, 0, &st[2]);
  ldg(wh, bn, 0, &st[4]); ldg(wl, bn, 0, &st[6]);
  sts(&sXh[0][0][0], &st[0]); sts(&sXl[0][0][0], &st[2]);
  sts(&sWh[0][0][0], &st[4]); sts(&sWl[0][0][0], &st[6]);
  __syncthreads();

  const int NK = KDIM / BK;   // 128
  for (int kt = 0; kt < NK; ++kt) {
    const int cur = kt & 1;
    if (kt + 1 < NK) {        // stage next k-tile into registers
      ldg(xh, bm, kt + 1, &st[0]); ldg(xl, bm, kt + 1, &st[2]);
      ldg(wh, bn, kt + 1, &st[4]); ldg(wl, bn, kt + 1, &st[6]);
    }

    v16bf ah[4], al[4], bh[2], bl[2];
#pragma unroll
    for (int im = 0; im < 4; ++im) {
      ah[im] = lfragA(&sXh[cur][0][0], wm + im * 16);
      al[im] = lfragA(&sXl[cur][0][0], wm + im * 16);
    }
#pragma unroll
    for (int jn = 0; jn < 2; ++jn) {
      bh[jn] = lfragB(&sWh[cur][0][0], wn + jn * 16);
      bl[jn] = lfragB(&sWl[cur][0][0], wn + jn * 16);
    }
#pragma unroll
    for (int im = 0; im < 4; ++im)
#pragma unroll
      for (int jn = 0; jn < 2; ++jn) {
        acc[im][jn] = __builtin_amdgcn_wmma_f32_16x16x32_bf16(
            false, ah[im], false, bh[jn], (short)0, acc[im][jn], false, false);
        acc[im][jn] = __builtin_amdgcn_wmma_f32_16x16x32_bf16(
            false, ah[im], false, bl[jn], (short)0, acc[im][jn], false, false);
        acc[im][jn] = __builtin_amdgcn_wmma_f32_16x16x32_bf16(
            false, al[im], false, bh[jn], (short)0, acc[im][jn], false, false);
      }

    if (kt + 1 < NK) {        // write next tile into the other buffer
      const int nxt = cur ^ 1;
      sts(&sXh[nxt][0][0], &st[0]); sts(&sXl[nxt][0][0], &st[2]);
      sts(&sWh[nxt][0][0], &st[4]); sts(&sWl[nxt][0][0], &st[6]);
    }
    __syncthreads();
  }

  // Epilogue: C/D layout (ISA 7.12.2): VGPR r, lanes 0-15 -> M=r, lanes 16-31
  // -> M=8+r; N = lane%16.
#pragma unroll
  for (int im = 0; im < 4; ++im)
#pragma unroll
    for (int jn = 0; jn < 2; ++jn) {
      const int n = bn + wn + jn * 16 + lr;
#pragma unroll
      for (int r = 0; r < 8; ++r) {
        const int m = bm + wm + im * 16 + r + 8 * kh;
        out[(size_t)m * NDIM + n] = acc[im][jn][r];
      }
    }
}

// ---------------------------------------------------------------------------
// Host launcher.  Workspace layout (bf16 planes):
//   wh [N*K] | wl [N*K] | xh [M*K] | xl [M*K]   -> ~192 MiB total
// ---------------------------------------------------------------------------
extern "C" void kernel_launch(void* const* d_in, const int* in_sizes, int n_in,
                              void* d_out, int out_size, void* d_ws,
                              size_t ws_size, hipStream_t stream) {
  const float* x  = (const float*)d_in[0];
  const float* Wb = (const float*)d_in[1];
  const float* A  = (const float*)d_in[2];
  const float* Bm = (const float*)d_in[3];
  float* out = (float*)d_out;

  __bf16* wh = (__bf16*)d_ws;
  __bf16* wl = wh + (size_t)NDIM * KDIM;
  __bf16* xh = wl + (size_t)NDIM * KDIM;
  __bf16* xl = xh + (size_t)MDIM * KDIM;

  dim3 gw(KDIM / 256, NDIM / 16);
  qlora_prep_w<<<gw, 256, 0, stream>>>(Wb, A, Bm, wh, wl);

  const int nblk_x = (int)(((size_t)MDIM * KDIM) / (256 * 8));
  qlora_prep_x<<<nblk_x, 256, 0, stream>>>(x, xh, xl);

  dim3 gg(NDIM / BN, MDIM / BM);   // 32 x 64 blocks
  qlora_gemm<<<gg, 256, 0, stream>>>(xh, xl, wh, wl, out);
}
